// GlmAsrAttention_20882130993355
// MI455X (gfx1250) — matrix operand order repaired and checked by reading, not compiled, not run on hardware
//
#include <hip/hip_runtime.h>
#include <hip/hip_bf16.h>

// ---------------------------------------------------------------------------
// GLM-ASR attention block for gfx1250 (MI455X), bf16 WMMA pipeline.
// Shapes: B=2, S=2048, HID=2048, H=16, HK=4, D=128, G=4 (GQA), non-causal.
// GEMMs: 32x64 wave tiles (8 wmma / 12 b128-loads per 32-deep K step).
// Attention: fused flash, K/V tiles in LDS shared by the 4 GQA heads,
// next-tile global_prefetch_b8 overlapped with compute.
// ---------------------------------------------------------------------------

typedef __bf16 bf16;
typedef __attribute__((ext_vector_type(16))) __bf16 v16bf;
typedef __attribute__((ext_vector_type(8)))  __bf16 bf16x8;
typedef __attribute__((ext_vector_type(8)))  float  v8f;

union Frag { v16bf v; bf16x8 h[2]; };

#define DEVI __device__ __forceinline__

DEVI bf16x8 ld8(const bf16* p) { return *reinterpret_cast<const bf16x8*>(p); }

DEVI v8f wmma_bf16(v16bf a, v16bf b, v8f c) {
  return __builtin_amdgcn_wmma_f32_16x16x32_bf16(false, a, false, b, (short)0, c,
                                                 false, false);
}

constexpr int CB   = 2;      // batch
constexpr int CS   = 2048;   // seq
constexpr int CHID = 2048;
constexpr int CH   = 16;     // q heads
constexpr int CHK  = 4;      // kv heads
constexpr int CD   = 128;    // head dim
constexpr int CG   = CH / CHK;
constexpr int CM   = CB * CS;              // 4096 rows
constexpr int CNQKV = (CH + 2 * CHK) * CD; // 3072

// ---------------------------------------------------------------------------
// f32 -> bf16 elementwise convert
// ---------------------------------------------------------------------------
__global__ void k_f2bf(const float* __restrict__ x, bf16* __restrict__ y, int n) {
  int i = blockIdx.x * 256 + threadIdx.x;
  if (i < n) y[i] = (bf16)x[i];
}

// ---------------------------------------------------------------------------
// QKV GEMM (NT): C[m,n] = sum_k X[m,k] * W[n,k] + b[n]
// X: [4096,2048] bf16, W: [3072,2048] bf16. Wave tile = 32(M) x 64(N):
// 2 A-frags x 4 B-frags -> 8 WMMAs per K step, B reused across both M blocks.
// Store scattered: Q [B,H,S,D], K [B,HK,S,D], V transposed [B,HK,D,S], bf16.
// ---------------------------------------------------------------------------
__global__ void k_qkv_gemm(const bf16* __restrict__ X, const bf16* __restrict__ W,
                           const float* __restrict__ bqkv,
                           bf16* __restrict__ Qo, bf16* __restrict__ Ko,
                           bf16* __restrict__ VTo) {
  const int NT64 = CNQKV / 64;                 // 48
  const int wt = blockIdx.x * 4 + (threadIdx.x >> 5);
  const int mt = wt / NT64;                    // 0..127 (32-row blocks)
  const int nt = wt % NT64;
  const int lane = threadIdx.x & 31;
  const int ncol = lane & 15;
  const int hi   = lane >> 4;

  const bf16* xrow0 = X + ((size_t)(mt * 32 + ncol)) * CHID;      // A half 0
  const bf16* xrow1 = xrow0 + (size_t)16 * CHID;                  // A half 1

  v8f acc[2][4];
#pragma unroll
  for (int u = 0; u < 2; ++u)
#pragma unroll
    for (int j = 0; j < 4; ++j) acc[u][j] = v8f{0, 0, 0, 0, 0, 0, 0, 0};

  for (int k0 = 0; k0 < CHID; k0 += 32) {
    Frag a0, a1;
    a0.h[0] = ld8(xrow0 + k0 + 8 * hi);
    a0.h[1] = ld8(xrow0 + k0 + 16 + 8 * hi);
    a1.h[0] = ld8(xrow1 + k0 + 8 * hi);
    a1.h[1] = ld8(xrow1 + k0 + 16 + 8 * hi);
#pragma unroll
    for (int j = 0; j < 4; ++j) {
      const int n = nt * 64 + j * 16 + ncol;       // B: n = lane&15
      const bf16* wr = W + (size_t)n * CHID + k0 + 8 * hi;
      Frag bf;
      bf.h[0] = ld8(wr);
      bf.h[1] = ld8(wr + 16);
      acc[0][j] = wmma_bf16(a0.v, bf.v, acc[0][j]);
      acc[1][j] = wmma_bf16(a1.v, bf.v, acc[1][j]);
    }
  }

#pragma unroll
  for (int u = 0; u < 2; ++u) {
#pragma unroll
    for (int j = 0; j < 4; ++j) {
      const int n = nt * 64 + j * 16 + ncol;
      const float bias = bqkv[n];
#pragma unroll
      for (int r = 0; r < 8; ++r) {
        const int row = mt * 32 + u * 16 + r + 8 * hi;   // C: M = r + 8*hi
        const float val = acc[u][j][r] + bias;
        const int b = row / CS, s = row % CS;
        if (n < CH * CD) {
          const int h = n / CD, d = n % CD;
          Qo[((b * CH + h) * CS + s) * CD + d] = (bf16)val;
        } else if (n < (CH + CHK) * CD) {
          const int nk = n - CH * CD;
          const int h = nk / CD, d = nk % CD;
          Ko[((b * CHK + h) * CS + s) * CD + d] = (bf16)val;
        } else {
          const int nv = n - (CH + CHK) * CD;
          const int h = nv / CD, d = nv % CD;
          VTo[((b * CHK + h) * CD + d) * CS + s] = (bf16)val;   // V transposed
        }
      }
    }
  }
}

// ---------------------------------------------------------------------------
// RoPE on Q and K (in place, bf16 storage, f32 math).
// ---------------------------------------------------------------------------
__global__ void k_rope(bf16* __restrict__ Q, bf16* __restrict__ K,
                       const float* __restrict__ cosp, const float* __restrict__ sinp) {
  const int total = CB * (CH + CHK) * CS * (CD / 2);
  int i = blockIdx.x * 256 + threadIdx.x;
  if (i >= total) return;
  const int d2 = i & 63;
  const int s  = (i >> 6) % CS;
  const int hh = (i >> 6) / CS % (CH + CHK);
  const int b  = (i >> 6) / (CS * (CH + CHK));
  bf16* base = (hh < CH) ? (Q + ((b * CH + hh) * CS + s) * CD)
                         : (K + ((b * CHK + (hh - CH)) * CS + s) * CD);
  const float* cb = cosp + (b * CS + s) * CD;
  const float* sb = sinp + (b * CS + s) * CD;
  const float x1 = (float)base[d2];
  const float x2 = (float)base[d2 + 64];
  base[d2]      = (bf16)(x1 * cb[d2] - x2 * sb[d2]);
  base[d2 + 64] = (bf16)(x2 * cb[d2 + 64] + x1 * sb[d2 + 64]);
}

// ---------------------------------------------------------------------------
// Fused flash attention. Block = (b, kv-head, 16-row q tile); 4 waves, wave g
// owns query head kh*G+g. K/V tiles (32 keys) staged in LDS, shared by waves.
// Online softmax; P routed through per-wave LDS to A-fragment layout.
// Next tile prefetched with global_prefetch_b8 while computing.
// ---------------------------------------------------------------------------
__global__ void k_attn(const bf16* __restrict__ Q, const bf16* __restrict__ K,
                       const bf16* __restrict__ VT, bf16* __restrict__ AO) {
  __shared__ bf16 Ks[32 * 128];     // [p][d]
  __shared__ bf16 Vs[128 * 32];     // [d][p]  (V already transposed in global)
  __shared__ bf16 Ps[4][16 * 32];   // per-wave P tile [m][p]

  const int QT = CS / 16;                       // 128
  const int qt = blockIdx.x % QT;
  const int kh = (blockIdx.x / QT) % CHK;
  const int b  = blockIdx.x / (QT * CHK);
  const int wave = threadIdx.x >> 5;
  const int h = kh * CG + wave;
  const int lane = threadIdx.x & 31;
  const int ncol = lane & 15;
  const int hi   = lane >> 4;
  const int t = threadIdx.x;

  // Q fragments (persistent): 16 rows x 128 d -> 4 A-frags of K-depth 32
  const bf16* qrow = Q + ((b * CH + h) * CS + qt * 16 + ncol) * CD;
  Frag qa[4];
#pragma unroll
  for (int f = 0; f < 4; ++f) {
    qa[f].h[0] = ld8(qrow + 32 * f + 8 * hi);
    qa[f].h[1] = ld8(qrow + 32 * f + 16 + 8 * hi);
  }

  float mrow[8], lrow[8];
  v8f acc[8];
#pragma unroll
  for (int r = 0; r < 8; ++r) { mrow[r] = -3.0e38f; lrow[r] = 0.0f; }
#pragma unroll
  for (int j = 0; j < 8; ++j) acc[j] = v8f{0, 0, 0, 0, 0, 0, 0, 0};

  const bf16* Kg0 = K + ((size_t)(b * CHK + kh) * CS) * CD;
  const bf16* Vg0 = VT + ((size_t)(b * CHK + kh) * CD) * CS;
  const float scale = 0.08838834764831845f;     // 1/sqrt(128)

  for (int kt = 0; kt < CS / 32; ++kt) {
    __syncthreads();                            // prior tile fully consumed
    // K tile: 32 contiguous rows of 128 => contiguous 4096 bf16 span
    const bf16* Kg = Kg0 + (size_t)kt * 32 * CD;
#pragma unroll
    for (int i = 0; i < 4; ++i) {
      const int off = (i * 128 + t) * 8;
      *reinterpret_cast<bf16x8*>(Ks + off) = ld8(Kg + off);
    }
    // V^T tile: 128 rows (d), 32 cols (p) starting at kt*32
    const bf16* Vg = Vg0 + kt * 32;
#pragma unroll
    for (int i = 0; i < 4; ++i) {
      const int c = i * 128 + t;
      const int dr = c >> 2;
      const int cc = (c & 3) * 8;
      *reinterpret_cast<bf16x8*>(Vs + dr * 32 + cc) = ld8(Vg + dr * CS + cc);
    }
    // prefetch next iteration's tiles (global_prefetch_b8; no counters, no VGPR)
    if (kt + 1 < CS / 32) {
      __builtin_prefetch(Kg + 32 * CD + t * 64, 0, 3);          // 8KB K tile
      __builtin_prefetch(Vg0 + (kt + 1) * 32 + (t & 127) * CS, 0, 3); // V rows
    }
    __syncthreads();

    // scores: 16(q) x 32(p), two 16x16 C-frags, K-depth 128 in 4 steps
    v8f s0 = v8f{0, 0, 0, 0, 0, 0, 0, 0};
    v8f s1 = v8f{0, 0, 0, 0, 0, 0, 0, 0};
#pragma unroll
    for (int f = 0; f < 4; ++f) {
      Frag kb0, kb1;
      const bf16* kr0 = Ks + ncol * 128 + 32 * f + 8 * hi;        // p = ncol
      const bf16* kr1 = Ks + (16 + ncol) * 128 + 32 * f + 8 * hi; // p = 16+ncol
      kb0.h[0] = ld8(kr0); kb0.h[1] = ld8(kr0 + 16);
      kb1.h[0] = ld8(kr1); kb1.h[1] = ld8(kr1 + 16);
      s0 = wmma_bf16(qa[f].v, kb0.v, s0);
      s1 = wmma_bf16(qa[f].v, kb1.v, s1);
    }

    // online softmax (row M = r + 8*hi lives in a 16-lane half)
    float corr[8];
#pragma unroll
    for (int r = 0; r < 8; ++r) {
      float v0 = s0[r] * scale, v1 = s1[r] * scale;
      float mx = fmaxf(v0, v1);
#pragma unroll
      for (int o = 8; o >= 1; o >>= 1) mx = fmaxf(mx, __shfl_xor(mx, o, 32));
      const float mnew = fmaxf(mrow[r], mx);
      const float c = __expf(mrow[r] - mnew);
      const float p0 = __expf(v0 - mnew);
      const float p1 = __expf(v1 - mnew);
      float rs = p0 + p1;
#pragma unroll
      for (int o = 8; o >= 1; o >>= 1) rs += __shfl_xor(rs, o, 32);
      lrow[r] = lrow[r] * c + rs;
      mrow[r] = mnew;
      corr[r] = c;
      s0[r] = p0;
      s1[r] = p1;
    }
#pragma unroll
    for (int j = 0; j < 8; ++j)
#pragma unroll
      for (int r = 0; r < 8; ++r) acc[j][r] *= corr[r];

    // P: C-layout -> LDS -> A-fragment layout (bf16)
    bf16* Pw = Ps[wave];
#pragma unroll
    for (int r = 0; r < 8; ++r) {
      const int row = r + 8 * hi;
      Pw[row * 32 + ncol]      = (bf16)s0[r];
      Pw[row * 32 + 16 + ncol] = (bf16)s1[r];
    }
    __syncthreads();

    Frag pa;
    const bf16* pr = Pw + ncol * 32;
    pa.h[0] = ld8(pr + 8 * hi);
    pa.h[1] = ld8(pr + 16 + 8 * hi);
#pragma unroll
    for (int j = 0; j < 8; ++j) {
      Frag vb;
      const bf16* vr = Vs + (j * 16 + ncol) * 32 + 8 * hi;        // d = j*16+ncol
      vb.h[0] = ld8(vr); vb.h[1] = ld8(vr + 16);
      acc[j] = wmma_bf16(pa.v, vb.v, acc[j]);
    }
  }

  // epilogue: normalize, store bf16 to [B, S, H*D]
  float rinv[8];
#pragma unroll
  for (int r = 0; r < 8; ++r) rinv[r] = 1.0f / lrow[r];
#pragma unroll
  for (int j = 0; j < 8; ++j)
#pragma unroll
    for (int r = 0; r < 8; ++r) {
      const int srow = qt * 16 + r + 8 * hi;
      const int d = j * 16 + ncol;
      AO[(((size_t)b * CS + srow) * CH + h) * CD + d] = (bf16)(acc[j][r] * rinv[r]);
    }
}

// ---------------------------------------------------------------------------
// Output GEMM (NT): Y[m,n] = sum_k A[m,k]*Wo[n,k] + bo[n], Y f32 [B,S,HID]
// Wave tile 32x64, same structure as k_qkv_gemm.
// ---------------------------------------------------------------------------
__global__ void k_out_gemm(const bf16* __restrict__ A, const bf16* __restrict__ W,
                           const float* __restrict__ bo, float* __restrict__ Y) {
  const int NT64 = CHID / 64;                   // 32
  const int wt = blockIdx.x * 4 + (threadIdx.x >> 5);
  const int mt = wt / NT64;                     // 0..127 (32-row blocks)
  const int nt = wt % NT64;
  const int lane = threadIdx.x & 31;
  const int ncol = lane & 15;
  const int hi   = lane >> 4;

  const bf16* arow0 = A + ((size_t)(mt * 32 + ncol)) * CHID;
  const bf16* arow1 = arow0 + (size_t)16 * CHID;

  v8f acc[2][4];
#pragma unroll
  for (int u = 0; u < 2; ++u)
#pragma unroll
    for (int j = 0; j < 4; ++j) acc[u][j] = v8f{0, 0, 0, 0, 0, 0, 0, 0};

  for (int k0 = 0; k0 < CHID; k0 += 32) {
    Frag a0, a1;
    a0.h[0] = ld8(arow0 + k0 + 8 * hi);
    a0.h[1] = ld8(arow0 + k0 + 16 + 8 * hi);
    a1.h[0] = ld8(arow1 + k0 + 8 * hi);
    a1.h[1] = ld8(arow1 + k0 + 16 + 8 * hi);
#pragma unroll
    for (int j = 0; j < 4; ++j) {
      const int n = nt * 64 + j * 16 + ncol;
      const bf16* wr = W + (size_t)n * CHID + k0 + 8 * hi;
      Frag bf;
      bf.h[0] = ld8(wr);
      bf.h[1] = ld8(wr + 16);
      acc[0][j] = wmma_bf16(a0.v, bf.v, acc[0][j]);
      acc[1][j] = wmma_bf16(a1.v, bf.v, acc[1][j]);
    }
  }

#pragma unroll
  for (int u = 0; u < 2; ++u)
#pragma unroll
    for (int j = 0; j < 4; ++j) {
      const int n = nt * 64 + j * 16 + ncol;
      const float bias = bo[n];
#pragma unroll
      for (int r = 0; r < 8; ++r) {
        const int row = mt * 32 + u * 16 + r + 8 * hi;
        Y[(size_t)row * CHID + n] = acc[u][j][r] + bias;
      }
    }
}

// ---------------------------------------------------------------------------
// Launcher. d_in: hidden(0) cos(1) sin(2) w_qkv(3) b_qkv(4) w_o(5) b_o(6).
// Workspace (~76 MB): bf16 copies + intermediates.
// ---------------------------------------------------------------------------
extern "C" void kernel_launch(void* const* d_in, const int* in_sizes, int n_in,
                              void* d_out, int out_size, void* d_ws, size_t ws_size,
                              hipStream_t stream) {
  const float* hs   = (const float*)d_in[0];
  const float* cosp = (const float*)d_in[1];
  const float* sinp = (const float*)d_in[2];
  const float* wqkv = (const float*)d_in[3];
  const float* bqkv = (const float*)d_in[4];
  const float* wo   = (const float*)d_in[5];
  const float* bo   = (const float*)d_in[6];
  float* out = (float*)d_out;

  char* ws = (char*)d_ws;
  const size_t nX  = (size_t)CM * CHID;        // 8,388,608
  const size_t nWq = (size_t)CNQKV * CHID;     // 6,291,456
  const size_t nWo = (size_t)CHID * CHID;      // 4,194,304
  const size_t nQ  = (size_t)CB * CH * CS * CD;
  const size_t nK  = (size_t)CB * CHK * CS * CD;

  bf16* Xbf  = (bf16*)ws;            ws += nX * 2;
  bf16* Wqbf = (bf16*)ws;            ws += nWq * 2;
  bf16* Wobf = (bf16*)ws;            ws += nWo * 2;
  bf16* Qbf  = (bf16*)ws;            ws += nQ * 2;
  bf16* Kbf  = (bf16*)ws;            ws += nK * 2;
  bf16* VTbf = (bf16*)ws;            ws += nK * 2;
  bf16* AObf = (bf16*)ws;            ws += nX * 2;

  // 1. converts
  k_f2bf<<<(int)((nX + 255) / 256), 256, 0, stream>>>(hs, Xbf, (int)nX);
  k_f2bf<<<(int)((nWq + 255) / 256), 256, 0, stream>>>(wqkv, Wqbf, (int)nWq);
  k_f2bf<<<(int)((nWo + 255) / 256), 256, 0, stream>>>(wo, Wobf, (int)nWo);

  // 2. QKV GEMM: (4096/32)*(3072/64) = 6144 wave tiles, 4 waves/block
  k_qkv_gemm<<<6144 / 4, 128, 0, stream>>>(Xbf, Wqbf, bqkv, Qbf, Kbf, VTbf);

  // 3. RoPE: B*(H+HK)*S*64 pairs
  {
    const int total = CB * (CH + CHK) * CS * (CD / 2);
    k_rope<<<(total + 255) / 256, 256, 0, stream>>>(Qbf, Kbf, cosp, sinp);
  }

  // 4. attention: B*HK*(S/16) = 1024 blocks, 128 threads
  k_attn<<<CB * CHK * (CS / 16), 128, 0, stream>>>(Qbf, Kbf, VTbf, AObf);

  // 5. output GEMM: (4096/32)*(2048/64) = 4096 wave tiles
  k_out_gemm<<<4096 / 4, 128, 0, stream>>>(AObf, Wobf, bo, out);
}
